// RGCN_Layer_39221641347105
// MI455X (gfx1250) — matrix operand order repaired and checked by reading, not compiled, not run on hardware
//
#include <hip/hip_runtime.h>

// Problem constants (match reference)
#define B_ 8
#define R_ 4
#define N_ 1024
#define D_ 256
#define L_ 2

typedef __attribute__((ext_vector_type(16))) __bf16 v16bf;
typedef __attribute__((ext_vector_type(8)))  __bf16 v8bf;
typedef __attribute__((ext_vector_type(8)))  float  v8f;
typedef __attribute__((ext_vector_type(4)))  int    v4i;

typedef __attribute__((address_space(1))) v4i* gv4i_ptr;
typedef __attribute__((address_space(3))) v4i* lv4i_ptr;

// LDS B-slab: 64 e-rows x 64 k, padded row stride of 72 elements.
// Row stride = 144 B = 36 banks; 16 consumer lanes hit 16 distinct bank
// quads (36*e mod 64 all distinct) -> conflict-free ds_load_b128.
#define LDSB_STRIDE 72
#define LDSB_TILE   (64 * LDSB_STRIDE)

// Zero-cost concatenation of two 4-VGPR halves into one 8-VGPR fragment.
__device__ __forceinline__ v16bf cat8(v8bf lo, v8bf hi) {
  return __builtin_shufflevector(lo, hi, 0, 1, 2, 3, 4, 5, 6, 7,
                                 8, 9, 10, 11, 12, 13, 14, 15);
}

// ---------------------------------------------------------------------------
// WMMA helper (CDNA5 wave32, 16x16x32 bf16 -> f32 accumulate)
// ---------------------------------------------------------------------------
__device__ __forceinline__ v8f wmma_bf16(v16bf a, v16bf b, v8f c) {
  return __builtin_amdgcn_wmma_f32_16x16x32_bf16(
      false, a, false, b, (short)0, c, false, false);
}

// A fragment: 16x32 bf16 tile, row-major source, leading dim lda.
// lanes 0-15: row=lane, K 0..15 (V0..7); lanes 16-31: K 16..31.
__device__ __forceinline__ v16bf load_frag_a(const __bf16* base, int lda) {
  int l    = threadIdx.x & 31;
  int row  = l & 15;
  int koff = (l >> 4) * 16;
  const __bf16* p = base + (size_t)row * lda + koff;
  return cat8(*reinterpret_cast<const v8bf*>(p),
              *reinterpret_cast<const v8bf*>(p + 8));
}

// B fragment from TRANSPOSED storage Bt[n][k] in global memory.
__device__ __forceinline__ v16bf load_frag_b(const __bf16* baseT, int ldb) {
  int l = threadIdx.x & 31;
  const __bf16* p = baseT + (size_t)(l & 15) * ldb + (l >> 4) * 16;
  return cat8(*reinterpret_cast<const v8bf*>(p),
              *reinterpret_cast<const v8bf*>(p + 8));
}

// B fragment from the padded LDS slab (Bt layout: [e][k], stride LDSB_STRIDE).
__device__ __forceinline__ v16bf load_frag_b_lds(const __bf16* baseT) {
  int l = threadIdx.x & 31;
  const __bf16* p = baseT + (l & 15) * LDSB_STRIDE + (l >> 4) * 16;
  return cat8(*reinterpret_cast<const v8bf*>(p),
              *reinterpret_cast<const v8bf*>(p + 8));
}

// ---------------------------------------------------------------------------
// Async global->LDS staging (GLOBAL_LOAD_ASYNC_TO_LDS_B128, ASYNCcnt).
// ---------------------------------------------------------------------------
__device__ __forceinline__ void stage16(const __bf16* src, __bf16* dst) {
#if defined(__has_builtin) && \
    __has_builtin(__builtin_amdgcn_global_load_async_to_lds_b128)
  __builtin_amdgcn_global_load_async_to_lds_b128(
      (gv4i_ptr)src, (lv4i_ptr)dst, 0, 0);
#else
  *reinterpret_cast<v8bf*>(dst) = *reinterpret_cast<const v8bf*>(src);
#endif
}

__device__ __forceinline__ void stage_wait() {
#if defined(__has_builtin) && __has_builtin(__builtin_amdgcn_s_wait_asynccnt)
  __builtin_amdgcn_s_wait_asynccnt(0);
#endif
  __syncthreads();
}

// Stage one 64e x 64k bf16 slab into `buf` (512 x 16B chunks, 2 per thread).
__device__ __forceinline__ void stage_slab(const __bf16* Bbase, int ld, int k0,
                                           __bf16* buf, int t) {
#pragma unroll
  for (int c = 0; c < 2; ++c) {
    int chunk = t * 2 + c;
    int er = chunk >> 3;          // 0..63
    int kc = (chunk & 7) * 8;     // 0..56
    stage16(Bbase + (size_t)er * ld + k0 + kc, buf + er * LDSB_STRIDE + kc);
  }
}

// ---------------------------------------------------------------------------
// Prep kernels
// ---------------------------------------------------------------------------
__global__ void cvt_bf16_kernel(const float* __restrict__ src,
                                __bf16* __restrict__ dst, int n) {
  int stride = gridDim.x * blockDim.x;
  for (int i = blockIdx.x * blockDim.x + threadIdx.x; i < n; i += stride)
    dst[i] = (__bf16)src[i];
}

// dst[mat][e][d] = bf16(src[mat][d][e])
__global__ void transpose_bf16_kernel(const float* __restrict__ src,
                                      __bf16* __restrict__ dst,
                                      int mats, int dim) {
  int total = mats * dim * dim;
  int stride = gridDim.x * blockDim.x;
  for (int idx = blockIdx.x * blockDim.x + threadIdx.x; idx < total;
       idx += stride) {
    int mat = idx / (dim * dim);
    int rem = idx - mat * dim * dim;
    int d = rem / dim;
    int e = rem - d * dim;
    dst[(size_t)mat * dim * dim + (size_t)e * dim + d] = (__bf16)src[idx];
  }
}

// denom[b][i] = 1 + sum_{r,j} adj[b][r][i][j]
__global__ void degree_kernel(const float* __restrict__ adj,
                              float* __restrict__ denom) {
  __shared__ float red[256];
  int bi = blockIdx.x;
  int b = bi >> 10;
  int i = bi & (N_ - 1);
  const float* base = adj + (size_t)b * R_ * N_ * N_ + (size_t)i * N_;
  float s = 0.0f;
  for (int t = threadIdx.x; t < R_ * N_; t += 256) {
    int r = t >> 10;
    int j = t & (N_ - 1);
    s += base[(size_t)r * N_ * N_ + j];
  }
  red[threadIdx.x] = s;
  __syncthreads();
  for (int w = 128; w > 0; w >>= 1) {
    if ((int)threadIdx.x < w) red[threadIdx.x] += red[threadIdx.x + w];
    __syncthreads();
  }
  if (threadIdx.x == 0) denom[bi] = red[0] + 1.0f;
}

// ---------------------------------------------------------------------------
// Per-relation linear: Yt[b][r][e][n] = bf16( x[b] @ Wr[r] + br[r] )
// One wave per 16x16 tile. grid = (B*R, N/16, D/16), block = 32.
// ---------------------------------------------------------------------------
__global__ void gemm_xw_kernel(const __bf16* __restrict__ xbf,
                               const __bf16* __restrict__ wrt,
                               const float*  __restrict__ br,
                               __bf16* __restrict__ yt) {
  int bri = blockIdx.x;
  int b = bri / R_;
  int r = bri - b * R_;
  int m0 = blockIdx.y * 16;
  int e0 = blockIdx.z * 16;

  const __bf16* xb = xbf + (size_t)b * N_ * D_ + (size_t)m0 * D_;
  const __bf16* wt = wrt + (size_t)r * D_ * D_ + (size_t)e0 * D_;

  v8f acc = {};
#pragma unroll
  for (int k0 = 0; k0 < D_; k0 += 32) {
    acc = wmma_bf16(load_frag_a(xb + k0, D_), load_frag_b(wt + k0, D_), acc);
  }

  int lane = threadIdx.x & 31;
  int e = e0 + (lane & 15);
  float bias = br[r * D_ + e];
  __bf16* yo = yt + (((size_t)b * R_ + r) * D_ + e) * N_ + m0 + (lane >> 4) * 8;
#pragma unroll
  for (int v = 0; v < 8; ++v) yo[v] = (__bf16)(acc[v] + bias);
}

// ---------------------------------------------------------------------------
// out[b][i][e] = relu( (sum_r adj[b,r] @ xW[b,r] + x[b]@W0 + b0) / denom )
// Block = 256 threads = 8 waves; block tile = 128(i) x 64(e).
// Each wave: one 16-row i-tile, 4 e-tiles register-blocked.
// B slab (64e x 64k bf16) double-buffered in LDS via async-to-LDS.
// grid = (B, N/128, D/64), block = 256.
// ---------------------------------------------------------------------------
__global__ void spmm_combine_kernel(const __bf16* __restrict__ adjbf,
                                    const __bf16* __restrict__ yt,
                                    const __bf16* __restrict__ xbf,
                                    const __bf16* __restrict__ w0t,
                                    const float*  __restrict__ b0,
                                    const float*  __restrict__ denom,
                                    float*  __restrict__ out,
                                    __bf16* __restrict__ xnext) {
  __shared__ __bf16 lds[2 * LDSB_TILE];   // 2 x 9216 B

  int b   = blockIdx.x;
  int i0b = blockIdx.y * 128;
  int e0  = blockIdx.z * 64;
  int wave = threadIdx.x >> 5;
  int i0w = i0b + wave * 16;
  int t = threadIdx.x;

  v8f acc[4] = {};

  // 5 segments: seg 0 = self term (K=256), segs 1..4 = relations (K=1024)
  for (int seg = 0; seg < 5; ++seg) {
    const __bf16* Abase;
    const __bf16* Bbase;
    int ld, K;
    if (seg == 0) {
      Abase = xbf + ((size_t)b * N_ + i0w) * D_;
      Bbase = w0t + (size_t)e0 * D_;
      ld = D_;  K = D_;
    } else {
      int r = seg - 1;
      Abase = adjbf + (((size_t)b * R_ + r) * N_ + i0w) * N_;
      Bbase = yt + (((size_t)b * R_ + r) * D_ + e0) * N_;
      ld = N_;  K = N_;
    }
    int nst = K / 64;

    // Prologue: stage first slab into buffer 0.
    stage_slab(Bbase, ld, 0, lds, t);

    for (int ks = 0; ks < nst; ++ks) {
      // Current slab ready + all waves done reading the other buffer.
      stage_wait();

      // Prefetch next slab into the other buffer (overlaps with compute).
      if (ks + 1 < nst)
        stage_slab(Bbase, ld, (ks + 1) * 64, lds + ((ks + 1) & 1) * LDSB_TILE, t);

      const __bf16* buf = lds + (ks & 1) * LDSB_TILE;
      int k0 = ks * 64;
      v16bf a0 = load_frag_a(Abase + k0, ld);
      v16bf a1 = load_frag_a(Abase + k0 + 32, ld);
#pragma unroll
      for (int et = 0; et < 4; ++et) {
        acc[et] = wmma_bf16(a0, load_frag_b_lds(buf + et * 16 * LDSB_STRIDE),
                            acc[et]);
        acc[et] = wmma_bf16(a1, load_frag_b_lds(buf + et * 16 * LDSB_STRIDE + 32),
                            acc[et]);
      }
    }
    __syncthreads();   // protect buffers before next segment's prologue
  }

  // --- epilogue: bias, normalize, relu, dual store (f32 out + bf16 next-x) ---
  int lane = threadIdx.x & 31;
#pragma unroll
  for (int et = 0; et < 4; ++et) {
    int e = e0 + et * 16 + (lane & 15);
    float bias = b0[e];
#pragma unroll
    for (int v = 0; v < 8; ++v) {
      int m = i0w + (lane >> 4) * 8 + v;
      float val = (acc[et][v] + bias) / denom[b * N_ + m];
      val = val > 0.0f ? val : 0.0f;
      size_t o = ((size_t)b * N_ + m) * D_ + e;
      out[o] = val;
      xnext[o] = (__bf16)val;
    }
  }
}

// ---------------------------------------------------------------------------
// Host launcher
// ---------------------------------------------------------------------------
extern "C" void kernel_launch(void* const* d_in, const int* in_sizes, int n_in,
                              void* d_out, int out_size, void* d_ws,
                              size_t ws_size, hipStream_t stream) {
  const float* nodes = (const float*)d_in[0];
  const float* adj   = (const float*)d_in[1];
  const float* Wr    = (const float*)d_in[2];
  const float* br    = (const float*)d_in[3];
  const float* W0    = (const float*)d_in[4];
  const float* b0    = (const float*)d_in[5];
  float* out = (float*)d_out;

  char* ws = (char*)d_ws;
  size_t off = 0;
  auto carve = [&](size_t bytes) {
    size_t o = off;
    off = (off + bytes + 255) & ~(size_t)255;
    return o;
  };
  const size_t adj_elems = (size_t)B_ * R_ * N_ * N_;
  const size_t x_elems   = (size_t)B_ * N_ * D_;
  const size_t y_elems   = (size_t)B_ * R_ * D_ * N_;
  const size_t wr_elems  = (size_t)L_ * R_ * D_ * D_;
  const size_t w0_elems  = (size_t)L_ * D_ * D_;

  __bf16* adjbf = (__bf16*)(ws + carve(adj_elems * 2));
  __bf16* ybf   = (__bf16*)(ws + carve(y_elems * 2));
  __bf16* xbf0  = (__bf16*)(ws + carve(x_elems * 2));
  __bf16* xbf1  = (__bf16*)(ws + carve(x_elems * 2));
  __bf16* wrt   = (__bf16*)(ws + carve(wr_elems * 2));
  __bf16* w0t   = (__bf16*)(ws + carve(w0_elems * 2));
  float*  denom = (float*)(ws + carve((size_t)B_ * N_ * 4));
  (void)ws_size; (void)in_sizes; (void)n_in; (void)out_size;

  // --- prep ---
  cvt_bf16_kernel<<<8192, 256, 0, stream>>>(adj, adjbf, (int)adj_elems);
  cvt_bf16_kernel<<<2048, 256, 0, stream>>>(nodes, xbf0, (int)x_elems);
  transpose_bf16_kernel<<<1024, 256, 0, stream>>>(Wr, wrt, L_ * R_, D_);
  transpose_bf16_kernel<<<512, 256, 0, stream>>>(W0, w0t, L_, D_);
  degree_kernel<<<B_ * N_, 256, 0, stream>>>(adj, denom);

  // --- layers ---
  __bf16* xcur = xbf0;
  __bf16* xnxt = xbf1;
  for (int l = 0; l < L_; ++l) {
    const __bf16* wrt_l = wrt + (size_t)l * R_ * D_ * D_;
    const __bf16* w0t_l = w0t + (size_t)l * D_ * D_;
    const float* br_l = br + (size_t)l * R_ * D_;
    const float* b0_l = b0 + (size_t)l * D_;
    float* out_l = out + (size_t)l * B_ * N_ * D_;

    gemm_xw_kernel<<<dim3(B_ * R_, N_ / 16, D_ / 16), 32, 0, stream>>>(
        xcur, wrt_l, br_l, ybf);
    spmm_combine_kernel<<<dim3(B_, N_ / 128, D_ / 64), 256, 0, stream>>>(
        adjbf, ybf, xcur, w0t_l, b0_l, denom, out_l, xnxt);

    __bf16* t = xcur; xcur = xnxt; xnxt = t;
  }
}
